// GCN_90589450207858
// MI455X (gfx1250) — matrix operand order repaired and checked by reading, not compiled, not run on hardware
//
#include <hip/hip_runtime.h>
#include <hip/hip_bf16.h>

typedef __attribute__((ext_vector_type(16))) _Float16 v16h;
typedef __attribute__((ext_vector_type(8)))  float    v8f;

#define WAVES_PER_BLOCK 8   // 256 threads = 8 wave32 on CDNA5

// ---------------------------------------------------------------------------
// WMMA GEMM:  OUT[n x Fo] = act(A[n x K]) @ W[K x Fo] (+bias, +relu epilogue)
//   - K, NTILES, RELU_IN, BIAS, RELU_OUT compile-time -> no predicated loads
//   - W staged via CDNA5 async copy: global_load_async_to_lds_b128 (ASYNCcnt),
//     then an LDS->LDS transpose + f32->f16 convert pass into sW.
//   - A fragments: one contiguous 16-float run per lane -> global_load_b128 x4
//   - B fragment = one 32B contiguous LDS read (2x ds_load_b128)
//   - requires n % 16 == 0 (true here: N=50000, G=512)
// A 16-bit 16x32 layout: lane(l<16) M=l holds K {0..7,16..23}; lanes 16-31: +8.
// B as used here (contiguous k, same order as the A loads): lanes 0-15 N=lane, K 0..15; lanes 16-31 K 16..31.
// D f32 16x16 layout: VGPR r -> M = r + 8*(lane>=16), N = lane&15.
// ---------------------------------------------------------------------------
template <int K, int NTILES, bool RELU_IN, bool BIAS, bool RELU_OUT>
__global__ void wmma_gemm_kernel(const float* __restrict__ A,
                                 const float* __restrict__ W,
                                 const float* __restrict__ bias,
                                 float* __restrict__ out, int n)
{
    constexpr int Fo     = NTILES * 16;
    constexpr int KSTEPS = (K + 31) / 32;
    constexpr int KPAD   = KSTEPS * 32;

    __shared__ __align__(16) float    sWf[K * Fo];      // raw f32 W (row-major)
    __shared__ __align__(16) _Float16 sW [Fo * KPAD];   // sW[col*KPAD+k] = f16 W[k][col]

    // ---- stage 1: async DMA of W (contiguous) into LDS, no VGPR round-trip ----
    // K*Fo*4 bytes is always a multiple of 16 here (Fo multiple of 16).
    {
        constexpr int CHUNKS = (K * Fo * 4) / 16;       // 16-byte chunks
        const unsigned ldsBase = (unsigned)(uintptr_t)&sWf[0];
        for (int c = threadIdx.x; c < CHUNKS; c += blockDim.x) {
            const unsigned    ldst = ldsBase + c * 16;
            const void* gsrc = (const char*)W + c * 16;
            asm volatile("global_load_async_to_lds_b128 %0, %1, off"
                         :: "v"(ldst), "v"(gsrc) : "memory");
        }
        asm volatile("s_wait_asynccnt 0" ::: "memory");
    }
    __syncthreads();

    // ---- stage 2: LDS->LDS transpose + f16 convert (zero-pad k >= K) ----
    for (int i = threadIdx.x; i < Fo * KPAD; i += blockDim.x) {
        const int col = i / KPAD;
        const int k   = i - col * KPAD;
        float v = 0.0f;
        if (k < K) v = sWf[k * Fo + col];               // compile-time folded if K==KPAD
        sW[i] = (_Float16)v;
    }
    __syncthreads();

    const int lane   = threadIdx.x & 31;
    const int waveId = threadIdx.x >> 5;
    const int mtile  = blockIdx.x * WAVES_PER_BLOCK + waveId;
    if (mtile * 16 >= n) return;               // uniform per-wave exit

    const int half = lane >> 4;                // 0: lanes 0-15, 1: lanes 16-31
    const int l15  = lane & 15;
    const int row  = mtile * 16 + l15;
    const float* __restrict__ abase = A + (size_t)row * K;

    v8f acc[NTILES] = {};

#pragma unroll
    for (int ks = 0; ks < KSTEPS; ++ks) {
        const int kbase = ks * 32;

        // ---- A fragment ----
        v16h a;
        if constexpr (K % 32 == 0) {
            const float* p = abase + kbase + (half << 4);
            const float4 q0 = *(const float4*)(p);
            const float4 q1 = *(const float4*)(p + 4);
            const float4 q2 = *(const float4*)(p + 8);
            const float4 q3 = *(const float4*)(p + 12);
            const float f[16] = {q0.x, q0.y, q0.z, q0.w, q1.x, q1.y, q1.z, q1.w,
                                 q2.x, q2.y, q2.z, q2.w, q3.x, q3.y, q3.z, q3.w};
#pragma unroll
            for (int i = 0; i < 16; ++i) {
                float v = f[i];
                if constexpr (RELU_IN) v = fmaxf(v, 0.0f);
                a[i] = (_Float16)v;
            }
        } else {
            // ragged K (layer 1, K=11): clamped unconditional loads + select
#pragma unroll
            for (int i = 0; i < 16; ++i) {
                const int k  = kbase + i + (half << 4);
                const int kc = (k < K) ? k : (K - 1);
                float v = abase[kc];
                v = (k < K) ? v : 0.0f;
                if constexpr (RELU_IN) v = fmaxf(v, 0.0f);
                a[i] = (_Float16)v;
            }
        }

        // ---- B fragments from LDS, one WMMA per N tile ----
#pragma unroll
        for (int t = 0; t < NTILES; ++t) {
            const int col = t * 16 + l15;
            const v16h b = *(const v16h*)&sW[col * KPAD + kbase + (half << 4)];
            acc[t] = __builtin_amdgcn_wmma_f32_16x16x32_f16(
                false, a, false, b, (short)0, acc[t], false, false);
        }
    }

    // ---- Epilogue: optional bias + ReLU, store f32 ----
#pragma unroll
    for (int t = 0; t < NTILES; ++t) {
        const int col = t * 16 + l15;
        float bv = 0.0f;
        if constexpr (BIAS) bv = bias[col];
#pragma unroll
        for (int r = 0; r < 8; ++r) {
            const int orow = mtile * 16 + r + (half << 3);
            float v = acc[t][r];
            if constexpr (BIAS) v += bv;
            if constexpr (RELU_OUT) v = fmaxf(v, 0.0f);
            out[(size_t)orow * Fo + col] = v;
        }
    }
}

// ---------------------------------------------------------------------------
// Elementwise / scatter helpers
// ---------------------------------------------------------------------------
__global__ void fill_kernel(float* __restrict__ p, float v, int n) {
    int i = blockIdx.x * blockDim.x + threadIdx.x;
    if (i < n) p[i] = v;
}

__global__ void degree_kernel(const int* __restrict__ dst, float* __restrict__ deg, int E) {
    int e = blockIdx.x * blockDim.x + threadIdx.x;
    if (e < E) atomicAdd(&deg[dst[e]], 1.0f);
}

__global__ void rsqrt_kernel(const float* __restrict__ deg, float* __restrict__ dis, int n) {
    int i = blockIdx.x * blockDim.x + threadIdx.x;
    if (i < n) dis[i] = rsqrtf(deg[i]);
}

__global__ void enorm_kernel(const int* __restrict__ src, const int* __restrict__ dst,
                             const float* __restrict__ dis, float* __restrict__ enm, int E) {
    int e = blockIdx.x * blockDim.x + threadIdx.x;
    if (e < E) enm[e] = dis[src[e]] * dis[dst[e]];
}

// agg = hw * dis^2 (self-loop) + bias  — base value before edge accumulation
__global__ void base_kernel(const float* __restrict__ hw, const float* __restrict__ dis,
                            const float* __restrict__ bias, float* __restrict__ agg,
                            int total, int foShift) {
    int i = blockIdx.x * blockDim.x + threadIdx.x;
    if (i >= total) return;
    int row = i >> foShift;
    int f   = i & ((1 << foShift) - 1);
    float d = dis[row];
    agg[i] = hw[i] * d * d + bias[f];
}

// agg[dst] += hw[src] * edge_norm   (one thread per edge-feature; coalesced per edge)
__global__ void scatter_kernel(const int* __restrict__ src, const int* __restrict__ dst,
                               const float* __restrict__ enm, const float* __restrict__ hw,
                               float* __restrict__ agg, int total, int foShift) {
    int i = blockIdx.x * blockDim.x + threadIdx.x;
    if (i >= total) return;
    int e  = i >> foShift;
    int f  = i & ((1 << foShift) - 1);
    int Fo = 1 << foShift;
    atomicAdd(&agg[(size_t)dst[e] * Fo + f], hw[(size_t)src[e] * Fo + f] * enm[e]);
}

__global__ void pool_sum_kernel(const float* __restrict__ h, const int* __restrict__ batch,
                                float* __restrict__ pooled, int total, int foShift) {
    int i = blockIdx.x * blockDim.x + threadIdx.x;
    if (i >= total) return;
    int row = i >> foShift;
    int f   = i & ((1 << foShift) - 1);
    atomicAdd(&pooled[(batch[row] << foShift) + f], h[i]);
}

__global__ void count_kernel(const int* __restrict__ batch, float* __restrict__ counts, int n) {
    int i = blockIdx.x * blockDim.x + threadIdx.x;
    if (i < n) atomicAdd(&counts[batch[i]], 1.0f);
}

__global__ void pool_div_kernel(float* __restrict__ pooled, const float* __restrict__ counts,
                                int total, int foShift) {
    int i = blockIdx.x * blockDim.x + threadIdx.x;
    if (i < total) pooled[i] /= fmaxf(counts[i >> foShift], 1.0f);
}

// out[g,t] = sum_f Hm[g,f] * fw2[f,t] + fb2[t]   (tiny: G x TGT)
__global__ void final_kernel(const float* __restrict__ Hm, const float* __restrict__ fw2,
                             const float* __restrict__ fb2, float* __restrict__ out,
                             int total, int TGT, int FM) {
    int i = blockIdx.x * blockDim.x + threadIdx.x;
    if (i >= total) return;
    int g = i / TGT;
    int t = i - g * TGT;
    float s = fb2[t];
    for (int f = 0; f < FM; ++f) s += Hm[g * FM + f] * fw2[f * TGT + t];
    out[i] = s;
}

// ---------------------------------------------------------------------------
// Launch
// ---------------------------------------------------------------------------
extern "C" void kernel_launch(void* const* d_in, const int* in_sizes, int n_in,
                              void* d_out, int out_size, void* d_ws, size_t ws_size,
                              hipStream_t stream) {
    const float* x     = (const float*)d_in[0];
    const int*   eidx  = (const int*)  d_in[1];
    const int*   batch = (const int*)  d_in[2];
    const float* W1 = (const float*)d_in[3];  const float* b1  = (const float*)d_in[4];
    const float* W2 = (const float*)d_in[5];  const float* b2  = (const float*)d_in[6];
    const float* W3 = (const float*)d_in[7];  const float* b3  = (const float*)d_in[8];
    const float* fw1 = (const float*)d_in[9]; const float* fb1 = (const float*)d_in[10];
    const float* fw2 = (const float*)d_in[11];const float* fb2 = (const float*)d_in[12];

    const int F1  = in_sizes[4];            // 32
    const int F2  = in_sizes[6];            // 64
    const int F3  = in_sizes[8];            // 128
    const int FM  = in_sizes[10];           // 64
    const int TGT = in_sizes[12];           // 12
    const int N   = in_sizes[2];            // 50000  (multiple of 16)
    const int E   = in_sizes[1] / 2;        // 600000
    const int G   = out_size / TGT;         // 512    (multiple of 16)

    const int* src = eidx;
    const int* dst = eidx + E;

    // Workspace layout (floats): ~54 MB total
    float* ws     = (float*)d_ws;
    float* deg    = ws;  ws += N;
    float* dis    = ws;  ws += N;
    float* enm    = ws;  ws += E;
    float* B0     = ws;  ws += (size_t)N * F3;     // HW buffer
    float* B1     = ws;  ws += (size_t)N * F3;     // AGG / layer output buffer
    float* pooled = ws;  ws += (size_t)G * F3;
    float* counts = ws;  ws += G;
    float* Hm     = ws;  ws += (size_t)G * FM;

    auto cdiv = [](int a, int b) { return (a + b - 1) / b; };
    const int T = 256;

    // --- normalization: deg = 1 + in-degree; dis = deg^-1/2; edge norms ---
    fill_kernel  <<<cdiv(N, T), T, 0, stream>>>(deg, 1.0f, N);
    degree_kernel<<<cdiv(E, T), T, 0, stream>>>(dst, deg, E);
    rsqrt_kernel <<<cdiv(N, T), T, 0, stream>>>(deg, dis, N);
    enorm_kernel <<<cdiv(E, T), T, 0, stream>>>(src, dst, dis, enm, E);

    const int mblocks = cdiv(cdiv(N, 16), WAVES_PER_BLOCK);

    // --- GCN layer 1: 11 -> 32 ---
    wmma_gemm_kernel<11, 2, false, false, false>
        <<<mblocks, 256, 0, stream>>>(x, W1, nullptr, B0, N);
    base_kernel   <<<cdiv(N * F1, T), T, 0, stream>>>(B0, dis, b1, B1, N * F1, 5);
    scatter_kernel<<<cdiv(E * F1, T), T, 0, stream>>>(src, dst, enm, B0, B1, E * F1, 5);

    // --- GCN layer 2: 32 -> 64 (ReLU fused into A-load) ---
    wmma_gemm_kernel<32, 4, true, false, false>
        <<<mblocks, 256, 0, stream>>>(B1, W2, nullptr, B0, N);
    base_kernel   <<<cdiv(N * F2, T), T, 0, stream>>>(B0, dis, b2, B1, N * F2, 6);
    scatter_kernel<<<cdiv(E * F2, T), T, 0, stream>>>(src, dst, enm, B0, B1, E * F2, 6);

    // --- GCN layer 3: 64 -> 128 (ReLU fused into A-load, no output ReLU) ---
    wmma_gemm_kernel<64, 8, true, false, false>
        <<<mblocks, 256, 0, stream>>>(B1, W3, nullptr, B0, N);
    base_kernel   <<<cdiv(N * F3, T), T, 0, stream>>>(B0, dis, b3, B1, N * F3, 7);
    scatter_kernel<<<cdiv(E * F3, T), T, 0, stream>>>(src, dst, enm, B0, B1, E * F3, 7);

    // --- global mean pool ---
    fill_kernel    <<<cdiv(G * F3, T), T, 0, stream>>>(pooled, 0.0f, G * F3);
    fill_kernel    <<<cdiv(G, T),      T, 0, stream>>>(counts, 0.0f, G);
    pool_sum_kernel<<<cdiv(N * F3, T), T, 0, stream>>>(B1, batch, pooled, N * F3, 7);
    count_kernel   <<<cdiv(N, T),      T, 0, stream>>>(batch, counts, N);
    pool_div_kernel<<<cdiv(G * F3, T), T, 0, stream>>>(pooled, counts, G * F3, 7);

    // --- MLP head: 128 -> 64 (bias+ReLU in WMMA epilogue) -> 12 (scalar) ---
    const int gblocks = cdiv(cdiv(G, 16), WAVES_PER_BLOCK);
    wmma_gemm_kernel<128, 4, false, true, true>
        <<<gblocks, 256, 0, stream>>>(pooled, fw1, fb1, Hm, G);
    final_kernel<<<cdiv(G * TGT, T), T, 0, stream>>>(Hm, fw2, fb2, (float*)d_out,
                                                     G * TGT, TGT, FM);
}
